// TransformerBlock_25056839205562
// MI455X (gfx1250) — matrix operand order repaired and checked by reading, not compiled, not run on hardware
//
#include <hip/hip_runtime.h>
#include <math.h>

// ---------------------------------------------------------------------------
// MI455X / gfx1250, wave32. Heavy GEMMs use v_wmma_f32_16x16x32_bf16
// (f32 in -> bf16 staged in LDS -> f32 accumulate). LDS tiles are stored so
// fragment elements are k-contiguous => packed bf16x2 (b32) DS loads.
// GEMM is double-buffered: one barrier per K-step, global fetch for step k+1
// issued before the WMMAs of step k.
// ---------------------------------------------------------------------------

typedef __attribute__((ext_vector_type(16))) __bf16 v16bf;
typedef __attribute__((ext_vector_type(2)))  __bf16 bf16x2;
typedef __attribute__((ext_vector_type(8)))  float  v8f;

#define D_   1024
#define NH_  16
#define NKV_ 8
#define HD_  64
#define E_   8
#define B_   2
#define S_   1024
#define T_   2048   // B_*S_
#define FF_  4096
#define EPS_ 1e-6f

__device__ __forceinline__ v8f wmma_bf16(v16bf a, v16bf b, v8f c) {
  return __builtin_amdgcn_wmma_f32_16x16x32_bf16(false, a, false, b,
                                                 (short)0, c, false, false);
}

// --- fragment loaders ------------------------------------------------------
// A-frag (16x32): lane<16 row m=lane, K={0..7,16..23}; lane>=16 row m=lane-16,
// K={8..15,24..31}. Element pairs (2p,2p+1) are k-contiguous.

// A-frag from f32 (global), row-major (m,k).
__device__ __forceinline__ v16bf load_a_frag_f32(const float* src, int ld) {
  const int lane = threadIdx.x & 31;
  const int half = lane >> 4;
  const int m    = lane & 15;
  v16bf a;
#pragma unroll
  for (int e = 0; e < 16; ++e) {
    int k = (e & 7) + ((e >> 3) << 4) + (half << 3);
    a[e] = (__bf16)src[(size_t)m * ld + k];
  }
  return a;
}

// A-frag from bf16 LDS, row-major (m,k), k contiguous; ld even.
__device__ __forceinline__ v16bf load_a_frag_lds(const __bf16* src, int ld) {
  const int lane = threadIdx.x & 31;
  const int half = lane >> 4;
  const int m    = lane & 15;
  v16bf a;
#pragma unroll
  for (int p = 0; p < 8; ++p) {
    int kb = ((p & 3) << 1) + ((p >> 2) << 4) + (half << 3);   // even
    bf16x2 t = *(const bf16x2*)(src + (size_t)m * ld + kb);
    a[2 * p]     = t[0];
    a[2 * p + 1] = t[1];
  }
  return a;
}

// B-frag (32x16) from f32 global, K-major: B[k][n] = src[k*ld + n].
__device__ __forceinline__ v16bf load_b_frag_f32(const float* src, size_t ld) {
  const int lane = threadIdx.x & 31;
  const int half = lane >> 4;
  const int n    = lane & 15;
  v16bf b;
#pragma unroll
  for (int e = 0; e < 16; ++e) {
    int k = (e & 7) + ((e >> 3) << 4) + (half << 3);
    b[e] = (__bf16)src[(size_t)k * ld + n];
  }
  return b;
}

// B-frag from bf16 LDS stored N-major: B[k][n] = src[n*ld + k], k contiguous.
__device__ __forceinline__ v16bf load_bT_frag_lds(const __bf16* src, int ld) {
  const int lane = threadIdx.x & 31;
  const int half = lane >> 4;
  const int n    = lane & 15;
  v16bf b;
#pragma unroll
  for (int p = 0; p < 8; ++p) {
    int kb = ((p & 3) << 1) + ((p >> 2) << 4) + (half << 3);   // even
    bf16x2 t = *(const bf16x2*)(src + (size_t)n * ld + kb);
    b[2 * p]     = t[0];
    b[2 * p + 1] = t[1];
  }
  return b;
}

// ---------------------------------------------------------------------------
// RMSNorm over rows of length D_
// ---------------------------------------------------------------------------
__global__ void rmsnorm_kernel(const float* __restrict__ x,
                               const float* __restrict__ w,
                               float* __restrict__ out) {
  const int row = blockIdx.x;
  const float* xr = x + (size_t)row * D_;
  float* yr = out + (size_t)row * D_;
  float ss = 0.f;
  for (int i = threadIdx.x; i < D_; i += blockDim.x) {
    float v = xr[i];
    ss += v * v;
  }
  __shared__ float red[8];
#pragma unroll
  for (int m = 16; m >= 1; m >>= 1) ss += __shfl_xor(ss, m);
  if ((threadIdx.x & 31) == 0) red[threadIdx.x >> 5] = ss;
  __syncthreads();
  if (threadIdx.x < 8) {
    float v = red[threadIdx.x];
#pragma unroll
    for (int m = 4; m >= 1; m >>= 1) v += __shfl_xor(v, m);
    if (threadIdx.x == 0) red[0] = v;
  }
  __syncthreads();
  const float inv = rsqrtf(red[0] / (float)D_ + EPS_);
  for (int i = threadIdx.x; i < D_; i += blockDim.x)
    yr[i] = xr[i] * inv * w[i];
}

// ---------------------------------------------------------------------------
// WMMA GEMM: C[M,N] = A[M,K] @ B[K,N] (+ residual).
// WG = 256 (8 waves) -> tile 128(M) x 64(N); wave owns 32x32 (2x2 frags,
// 4 WMMA / K-step). Double-buffered bf16 LDS: A row-major, B transposed
// (N-major); both fragment types use packed b32 DS loads. One barrier/step.
// ---------------------------------------------------------------------------
__global__ void gemm_bf16_wmma(const float* __restrict__ A,
                               const float* __restrict__ Bm,
                               const float* __restrict__ resid,
                               float* __restrict__ C,
                               int M, int N, int K) {
  __shared__ __bf16 As[2][128 * 34];   // row m, k contiguous; dword stride 17
  __shared__ __bf16 Bs[2][64 * 34];    // row n, k contiguous
  const int tid  = threadIdx.x;
  const int wave = tid >> 5;
  const int wm   = wave & 3;           // 4 row tiles of 32
  const int wn   = wave >> 2;          // 2 col tiles of 32
  const int m0   = blockIdx.y * 128;
  const int n0   = blockIdx.x * 64;

  v8f acc[2][2] = {};
  float4 ar[4], br[2];                 // register-staged tile

  auto global_fetch = [&](int k0) {
#pragma unroll
    for (int j = 0; j < 4; ++j) {      // A: 128x32, 1024 float4
      int i = tid + j * 256;
      int row = i >> 3, c4 = (i & 7) << 2;
      ar[j] = *(const float4*)(A + (size_t)(m0 + row) * K + k0 + c4);
    }
#pragma unroll
    for (int j = 0; j < 2; ++j) {      // B: 32x64, 512 float4
      int i = tid + j * 256;
      int kr = i >> 4, c4 = (i & 15) << 2;
      br[j] = *(const float4*)(Bm + (size_t)(k0 + kr) * N + n0 + c4);
    }
  };
  auto lds_store = [&](int buf) {
#pragma unroll
    for (int j = 0; j < 4; ++j) {
      int i = tid + j * 256;
      int row = i >> 3, c4 = (i & 7) << 2;
      __bf16* dp = As[buf] + row * 34 + c4;
      dp[0] = (__bf16)ar[j].x; dp[1] = (__bf16)ar[j].y;
      dp[2] = (__bf16)ar[j].z; dp[3] = (__bf16)ar[j].w;
    }
#pragma unroll
    for (int j = 0; j < 2; ++j) {
      int i = tid + j * 256;
      int kr = i >> 4, c4 = (i & 15) << 2;
      Bs[buf][(c4 + 0) * 34 + kr] = (__bf16)br[j].x;
      Bs[buf][(c4 + 1) * 34 + kr] = (__bf16)br[j].y;
      Bs[buf][(c4 + 2) * 34 + kr] = (__bf16)br[j].z;
      Bs[buf][(c4 + 3) * 34 + kr] = (__bf16)br[j].w;
    }
  };

  global_fetch(0);
  lds_store(0);
  __syncthreads();

  const int nk = K >> 5;
  for (int kk = 0; kk < nk; ++kk) {
    const int cur = kk & 1;
    if (kk + 1 < nk) global_fetch((kk + 1) << 5);   // overlap with WMMAs
    v16bf a0 = load_a_frag_lds(As[cur] + (wm * 32) * 34, 34);
    v16bf a1 = load_a_frag_lds(As[cur] + (wm * 32 + 16) * 34, 34);
    v16bf b0 = load_bT_frag_lds(Bs[cur] + (wn * 32) * 34, 34);
    v16bf b1 = load_bT_frag_lds(Bs[cur] + (wn * 32 + 16) * 34, 34);
    acc[0][0] = wmma_bf16(a0, b0, acc[0][0]);
    acc[0][1] = wmma_bf16(a0, b1, acc[0][1]);
    acc[1][0] = wmma_bf16(a1, b0, acc[1][0]);
    acc[1][1] = wmma_bf16(a1, b1, acc[1][1]);
    if (kk + 1 < nk) lds_store(1 - cur);
    __syncthreads();
  }

  const int lane = tid & 31, half = lane >> 4, ln = lane & 15;
  if (resid != nullptr) {
#pragma unroll
    for (int sm = 0; sm < 2; ++sm)
#pragma unroll
      for (int sn = 0; sn < 2; ++sn) {
        const size_t mb = (size_t)(m0 + wm * 32 + sm * 16 + half * 8);
        const int cn = n0 + wn * 32 + sn * 16 + ln;
#pragma unroll
        for (int r = 0; r < 8; ++r) {
          size_t idx = (mb + r) * (size_t)N + cn;
          C[idx] = acc[sm][sn][r] + resid[idx];
        }
      }
  } else {
#pragma unroll
    for (int sm = 0; sm < 2; ++sm)
#pragma unroll
      for (int sn = 0; sn < 2; ++sn) {
        const size_t mb = (size_t)(m0 + wm * 32 + sm * 16 + half * 8);
        const int cn = n0 + wn * 32 + sn * 16 + ln;
#pragma unroll
        for (int r = 0; r < 8; ++r)
          C[(mb + r) * (size_t)N + cn] = acc[sm][sn][r];
      }
  }
}

// ---------------------------------------------------------------------------
// Per-head RMSNorm (HD_=64) + RoPE, in place, for q and k.
// ---------------------------------------------------------------------------
__global__ void qknorm_rope_kernel(float* __restrict__ qbuf,
                                   float* __restrict__ kbuf,
                                   const float* __restrict__ qw,
                                   const float* __restrict__ kw) {
  const int vid = blockIdx.x;
  const int d   = threadIdx.x;           // 0..63
  const int nq  = T_ * NH_;
  float* vec;
  const float* w;
  int token;
  if (vid < nq) {
    token = vid / NH_;
    vec = qbuf + (size_t)vid * HD_;
    w = qw;
  } else {
    int v2 = vid - nq;
    token = v2 / NKV_;
    vec = kbuf + (size_t)v2 * HD_;
    w = kw;
  }
  __shared__ float sv[64];
  __shared__ float sn[64];
  float x = vec[d];
  sv[d] = x * x;
  __syncthreads();
  for (int st = 32; st > 0; st >>= 1) {
    if (d < st) sv[d] += sv[d + st];
    __syncthreads();
  }
  const float inv = rsqrtf(sv[0] / (float)HD_ + EPS_);
  const float xn = x * inv * w[d];
  sn[d] = xn;
  __syncthreads();
  const float partner = (d < 32) ? sn[d + 32] : sn[d - 32];
  const float rot = (d < 32) ? -partner : partner;
  const int   i = d & 31;
  const float invf = __powf(1.0e6f, -(float)i / 32.f);
  const float pos = (float)(token & (S_ - 1));
  const float ang = pos * invf;
  vec[d] = xn * __cosf(ang) + rot * __sinf(ang);
}

// ---------------------------------------------------------------------------
// Causal flash attention. grid = (S/128, NH, B); 256 threads = 8 waves.
// K staged bf16 key-major (score B-frags k-contiguous), V staged bf16
// dim-major (AV B-frags k-contiguous). Staging loads batched into registers.
// ---------------------------------------------------------------------------
__global__ void attention_kernel(const float* __restrict__ qb,
                                 const float* __restrict__ kb,
                                 const float* __restrict__ vb,
                                 float* __restrict__ ob) {
  const int b = blockIdx.z, h = blockIdx.y, qblk = blockIdx.x;
  const int kvh = h >> 1;
  const int tid = threadIdx.x, wave = tid >> 5, lane = tid & 31;
  const int half = lane >> 4, ln = lane & 15;
  const int q0 = qblk * 128 + wave * 16;

  __shared__ __bf16 Ks[32 * 66];       // [key][hd]
  __shared__ __bf16 Vst[64 * 66];      // [dim][key]  (transposed)
  __shared__ __bf16 Ps[8][16 * 34];    // per-wave P tile [q][key]

  const float* qptr = qb + ((size_t)(b * S_ + q0) * NH_ + h) * HD_;
  const v16bf aq0 = load_a_frag_f32(qptr, NH_ * HD_);
  const v16bf aq1 = load_a_frag_f32(qptr + 32, NH_ * HD_);

  v8f oc0 = {}, oc1 = {}, oc2 = {}, oc3 = {};
  float mrow[8], lrow[8];
#pragma unroll
  for (int r = 0; r < 8; ++r) { mrow[r] = -1e30f; lrow[r] = 0.f; }

  const int nkb = (qblk + 1) * 4;
  for (int kb2 = 0; kb2 < nkb; ++kb2) {
    const int kbase = kb2 << 5;
    // batched K/V staging: 512 float4s, 2 per thread, loads issued first
    float4 kf[2], vf[2];
#pragma unroll
    for (int j = 0; j < 2; ++j) {
      int i = tid + j * 256;
      int r = i >> 4, c4 = (i & 15) << 2;
      size_t src = ((size_t)(b * S_ + kbase + r) * NKV_ + kvh) * HD_ + c4;
      kf[j] = *(const float4*)(kb + src);
      vf[j] = *(const float4*)(vb + src);
    }
#pragma unroll
    for (int j = 0; j < 2; ++j) {
      int i = tid + j * 256;
      int r = i >> 4, c4 = (i & 15) << 2;
      __bf16* kp = Ks + r * 66 + c4;
      kp[0] = (__bf16)kf[j].x; kp[1] = (__bf16)kf[j].y;
      kp[2] = (__bf16)kf[j].z; kp[3] = (__bf16)kf[j].w;
      Vst[(c4 + 0) * 66 + r] = (__bf16)vf[j].x;
      Vst[(c4 + 1) * 66 + r] = (__bf16)vf[j].y;
      Vst[(c4 + 2) * 66 + r] = (__bf16)vf[j].z;
      Vst[(c4 + 3) * 66 + r] = (__bf16)vf[j].w;
    }
    __syncthreads();

    if (kbase <= q0 + 15) {        // wave-uniform causal skip
      v8f s0 = {}, s1 = {};
      s0 = wmma_bf16(aq0, load_bT_frag_lds(Ks, 66), s0);
      s0 = wmma_bf16(aq1, load_bT_frag_lds(Ks + 32, 66), s0);
      s1 = wmma_bf16(aq0, load_bT_frag_lds(Ks + 16 * 66, 66), s1);
      s1 = wmma_bf16(aq1, load_bT_frag_lds(Ks + 16 * 66 + 32, 66), s1);

      const float sc = 0.125f;     // 1/sqrt(64)
      float nm[8];
#pragma unroll
      for (int r = 0; r < 8; ++r) {
        const int qg = q0 + r + half * 8;
        float v0 = (kbase + ln      <= qg) ? s0[r] * sc : -1e30f;
        float v1 = (kbase + 16 + ln <= qg) ? s1[r] * sc : -1e30f;
        s0[r] = v0; s1[r] = v1;
        nm[r] = fmaxf(v0, v1);
      }
#pragma unroll
      for (int r = 0; r < 8; ++r) {
#pragma unroll
        for (int msk = 8; msk >= 1; msk >>= 1)
          nm[r] = fmaxf(nm[r], __shfl_xor(nm[r], msk));
        const float mnew  = fmaxf(mrow[r], nm[r]);
        const float scale = __expf(mrow[r] - mnew);
        mrow[r] = mnew;
        const float p0 = __expf(s0[r] - mnew);
        const float p1 = __expf(s1[r] - mnew);
        float rs = p0 + p1;
#pragma unroll
        for (int msk = 8; msk >= 1; msk >>= 1) rs += __shfl_xor(rs, msk);
        lrow[r] = lrow[r] * scale + rs;
        oc0[r] *= scale; oc1[r] *= scale; oc2[r] *= scale; oc3[r] *= scale;
        Ps[wave][(r + half * 8) * 34 + ln]      = (__bf16)p0;
        Ps[wave][(r + half * 8) * 34 + 16 + ln] = (__bf16)p1;
      }
      asm volatile("s_wait_dscnt 0x0" ::: "memory");
      const v16bf pa = load_a_frag_lds(&Ps[wave][0], 34);
      oc0 = wmma_bf16(pa, load_bT_frag_lds(Vst + 0 * 66,  66), oc0);
      oc1 = wmma_bf16(pa, load_bT_frag_lds(Vst + 16 * 66, 66), oc1);
      oc2 = wmma_bf16(pa, load_bT_frag_lds(Vst + 32 * 66, 66), oc2);
      oc3 = wmma_bf16(pa, load_bT_frag_lds(Vst + 48 * 66, 66), oc3);
    }
    __syncthreads();
  }

  float* op = ob + ((size_t)(b * S_ + q0) * NH_ + h) * HD_;
#pragma unroll
  for (int r = 0; r < 8; ++r) {
    const float invl = 1.f / lrow[r];
    const size_t row = (size_t)(r + half * 8) * (NH_ * HD_);
    op[row + ln]      = oc0[r] * invl;
    op[row + 16 + ln] = oc1[r] * invl;
    op[row + 32 + ln] = oc2[r] * invl;
    op[row + 48 + ln] = oc3[r] * invl;
  }
}

// ---------------------------------------------------------------------------
// Router (N=8, too small for WMMA -> scalar), softmax/top-2/aux
// ---------------------------------------------------------------------------
__global__ void router_kernel(const float* __restrict__ t,
                              const float* __restrict__ rw,
                              float* __restrict__ logits) {
  const int idx = blockIdx.x * blockDim.x + threadIdx.x;
  if (idx >= T_ * E_) return;
  const int tok = idx >> 3, e = idx & 7;
  const float* tr = t + (size_t)tok * D_;
  float acc = 0.f;
  for (int k = 0; k < D_; ++k) acc += tr[k] * rw[(size_t)k * E_ + e];
  logits[idx] = acc;
}

__global__ void zero16_kernel(float* p) {
  if (threadIdx.x < 16) p[threadIdx.x] = 0.f;
}

__global__ void topk_kernel(const float* __restrict__ logits,
                            float* __restrict__ combine,
                            float* __restrict__ facc,
                            float* __restrict__ pacc) {
  const int tok = blockIdx.x * blockDim.x + threadIdx.x;
  if (tok >= T_) return;
  float l[E_], p[E_];
  float mx = -1e30f;
#pragma unroll
  for (int e = 0; e < E_; ++e) { l[e] = logits[tok * E_ + e]; mx = fmaxf(mx, l[e]); }
  float sum = 0.f;
#pragma unroll
  for (int e = 0; e < E_; ++e) { p[e] = __expf(l[e] - mx); sum += p[e]; }
  const float isum = 1.f / sum;
#pragma unroll
  for (int e = 0; e < E_; ++e) p[e] *= isum;
  int i0 = 0;
#pragma unroll
  for (int e = 1; e < E_; ++e) if (p[e] > p[i0]) i0 = e;
  int i1 = (i0 == 0) ? 1 : 0;
#pragma unroll
  for (int e = 0; e < E_; ++e) if (e != i0 && p[e] > p[i1]) i1 = e;
  const float s2 = p[i0] + p[i1];
#pragma unroll
  for (int e = 0; e < E_; ++e) combine[tok * E_ + e] = 0.f;
  combine[tok * E_ + i0] = p[i0] / s2;
  combine[tok * E_ + i1] = p[i1] / s2;
  unsafeAtomicAdd(&facc[i0], 1.f);
  unsafeAtomicAdd(&facc[i1], 1.f);
#pragma unroll
  for (int e = 0; e < E_; ++e) unsafeAtomicAdd(&pacc[e], p[e]);
}

__global__ void aux_kernel(const float* __restrict__ facc,
                           const float* __restrict__ pacc,
                           float* __restrict__ outp) {
  float a = 0.f;
  for (int e = 0; e < E_; ++e)
    a += (facc[e] / ((float)T_ * 2.f)) * (pacc[e] / (float)T_);
  *outp = (float)E_ * a;
}

// ---------------------------------------------------------------------------
// Fused MoE: one WG per (16-token tile, expert). t staged bf16 in LDS,
// gate/up share one A-frag across 2 N-tiles (1 A + 4 B + 4 WMMA / K-step),
// SiLU*up*combine kept as a 16x4096 bf16 activation tile in ~160 KB of the
// 320 KB WGP LDS, then the down GEMM reads A-frags straight from LDS and
// atomically accumulates into the output residual stream. Expert weights
// (402 MB total, streamed once) are the HBM-bound part -> prefetch.
// ---------------------------------------------------------------------------
#define ATILE_LD 1026
#define ACT_LD   4098
__global__ void moe_kernel(const float* __restrict__ t,
                           const float* __restrict__ combine,
                           const float* __restrict__ wg,
                           const float* __restrict__ wu,
                           const float* __restrict__ wd,
                           float* __restrict__ y) {
  extern __shared__ char smem[];
  __bf16* Atile = (__bf16*)smem;                    // [16][ATILE_LD]
  __bf16* Act   = Atile + 16 * ATILE_LD;            // [16][ACT_LD]
  __shared__ float cw[16];

  const int e    = blockIdx.y;
  const int t0   = blockIdx.x * 16;
  const int tid  = threadIdx.x, wave = tid >> 5;
  const int lane = tid & 31, half = lane >> 4, ln = lane & 15;

  if (tid < 16) cw[tid] = combine[(size_t)(t0 + tid) * E_ + e];
  __syncthreads();
  bool any = false;
#pragma unroll
  for (int i = 0; i < 16; ++i) any = any || (cw[i] != 0.f);
  if (!any) return;   // block-uniform early exit: no token routed here

  // stage t tile: 4096 float4s, batched 4 at a time per thread
  for (int base = 0; base < 16 * (D_ / 4); base += 1024) {
    float4 f[4];
#pragma unroll
    for (int j = 0; j < 4; ++j) {
      int i = base + tid + j * 256;
      int r = i >> 8, c4 = (i & 255) << 2;
      f[j] = *(const float4*)(t + (size_t)(t0 + r) * D_ + c4);
    }
#pragma unroll
    for (int j = 0; j < 4; ++j) {
      int i = base + tid + j * 256;
      int r = i >> 8, c4 = (i & 255) << 2;
      __bf16* dp = Atile + r * ATILE_LD + c4;
      dp[0] = (__bf16)f[j].x; dp[1] = (__bf16)f[j].y;
      dp[2] = (__bf16)f[j].z; dp[3] = (__bf16)f[j].w;
    }
  }
  __syncthreads();

  const float* wge = wg + (size_t)e * D_ * FF_;
  const float* wue = wu + (size_t)e * D_ * FF_;
  for (int nt = wave * 2; nt < FF_ / 16; nt += 16) {   // 2 N-tiles / iter
    v8f g0 = {}, u0 = {}, g1 = {}, u1 = {};
    for (int ks = 0; ks < D_ / 32; ++ks) {
      __builtin_prefetch(wge + (size_t)(ks * 32 + 64) * FF_ + nt * 16, 0, 1);
      v16bf a   = load_a_frag_lds(Atile + ks * 32, ATILE_LD);
      v16bf bg0 = load_b_frag_f32(wge + (size_t)(ks * 32) * FF_ + nt * 16, FF_);
      v16bf bu0 = load_b_frag_f32(wue + (size_t)(ks * 32) * FF_ + nt * 16, FF_);
      v16bf bg1 = load_b_frag_f32(wge + (size_t)(ks * 32) * FF_ + (nt + 1) * 16, FF_);
      v16bf bu1 = load_b_frag_f32(wue + (size_t)(ks * 32) * FF_ + (nt + 1) * 16, FF_);
      g0 = wmma_bf16(a, bg0, g0);
      u0 = wmma_bf16(a, bu0, u0);
      g1 = wmma_bf16(a, bg1, g1);
      u1 = wmma_bf16(a, bu1, u1);
    }
#pragma unroll
    for (int r = 0; r < 8; ++r) {
      const int row = r + half * 8;
      const float s0 = g0[r] / (1.f + __expf(-g0[r]));
      const float s1 = g1[r] / (1.f + __expf(-g1[r]));
      Act[row * ACT_LD + nt * 16 + ln]        = (__bf16)(cw[row] * s0 * u0[r]);
      Act[row * ACT_LD + (nt + 1) * 16 + ln]  = (__bf16)(cw[row] * s1 * u1[r]);
    }
  }
  __syncthreads();

  const float* wde = wd + (size_t)e * FF_ * D_;
  for (int nt = wave * 2; nt < D_ / 16; nt += 16) {    // 2 N-tiles / iter
    v8f a0 = {}, a1 = {};
    for (int ks = 0; ks < FF_ / 32; ++ks) {
      __builtin_prefetch(wde + (size_t)(ks * 32 + 64) * D_ + nt * 16, 0, 1);
      v16bf a   = load_a_frag_lds(Act + ks * 32, ACT_LD);
      v16bf bd0 = load_b_frag_f32(wde + (size_t)(ks * 32) * D_ + nt * 16, D_);
      v16bf bd1 = load_b_frag_f32(wde + (size_t)(ks * 32) * D_ + (nt + 1) * 16, D_);
      a0 = wmma_bf16(a, bd0, a0);
      a1 = wmma_bf16(a, bd1, a1);
    }
#pragma unroll
    for (int r = 0; r < 8; ++r) {
      const int row = r + half * 8;
      unsafeAtomicAdd(&y[(size_t)(t0 + row) * D_ + nt * 16 + ln], a0[r]);
      unsafeAtomicAdd(&y[(size_t)(t0 + row) * D_ + (nt + 1) * 16 + ln], a1[r]);
    }
  }
}

// ---------------------------------------------------------------------------
// Launch
// ---------------------------------------------------------------------------
extern "C" void kernel_launch(void* const* d_in, const int* in_sizes, int n_in,
                              void* d_out, int out_size, void* d_ws, size_t ws_size,
                              hipStream_t stream) {
  const float* x    = (const float*)d_in[0];
  const float* n1w  = (const float*)d_in[1];
  const float* n2w  = (const float*)d_in[2];
  const float* wq   = (const float*)d_in[3];
  const float* wk   = (const float*)d_in[4];
  const float* wv   = (const float*)d_in[5];
  const float* wo   = (const float*)d_in[6];
  const float* qnw  = (const float*)d_in[7];
  const float* knw  = (const float*)d_in[8];
  const float* rw   = (const float*)d_in[9];
  const float* wg   = (const float*)d_in[10];
  const float* wu   = (const float*)d_in[11];
  const float* wd   = (const float*)d_in[12];
  float* out = (float*)d_out;
  float* ws  = (float*)d_ws;

  float* h       = ws;
  float* qbuf    = h    + (size_t)T_ * D_;
  float* kbuf    = qbuf + (size_t)T_ * D_;
  float* vbuf    = kbuf + (size_t)T_ * (NKV_*HD_);
  float* obuf    = vbuf + (size_t)T_ * (NKV_*HD_);
  float* tbuf    = obuf + (size_t)T_ * D_;
  float* logits  = tbuf + (size_t)T_ * D_;
  float* combine = logits + (size_t)T_ * E_;
  float* facc    = combine + (size_t)T_ * E_;
  float* pacc    = facc + 8;

  rmsnorm_kernel<<<T_, 256, 0, stream>>>(x, n1w, h);

  gemm_bf16_wmma<<<dim3(D_/64, T_/128), 256, 0, stream>>>(h, wq, nullptr, qbuf, T_, D_, D_);
  gemm_bf16_wmma<<<dim3((NKV_*HD_)/64, T_/128), 256, 0, stream>>>(h, wk, nullptr, kbuf, T_, NKV_*HD_, D_);
  gemm_bf16_wmma<<<dim3((NKV_*HD_)/64, T_/128), 256, 0, stream>>>(h, wv, nullptr, vbuf, T_, NKV_*HD_, D_);

  qknorm_rope_kernel<<<T_*(NH_+NKV_), 64, 0, stream>>>(qbuf, kbuf, qnw, knw);

  attention_kernel<<<dim3(S_/128, NH_, B_), 256, 0, stream>>>(qbuf, kbuf, vbuf, obuf);

  gemm_bf16_wmma<<<dim3(D_/64, T_/128), 256, 0, stream>>>(obuf, wo, x, out, T_, D_, D_);

  rmsnorm_kernel<<<T_, 256, 0, stream>>>(out, n2w, tbuf);
  router_kernel<<<(T_*E_ + 255)/256, 256, 0, stream>>>(tbuf, rw, logits);
  zero16_kernel<<<1, 32, 0, stream>>>(facc);
  topk_kernel<<<(T_ + 255)/256, 256, 0, stream>>>(logits, combine, facc, pacc);
  aux_kernel<<<1, 1, 0, stream>>>(facc, pacc, out + (size_t)T_ * D_);

  const size_t moe_lds = (size_t)16 * ATILE_LD * 2 + (size_t)16 * ACT_LD * 2;
  moe_kernel<<<dim3(T_/16, E_), 256, moe_lds, stream>>>(tbuf, combine, wg, wu, wd, out);

  (void)in_sizes; (void)n_in; (void)out_size; (void)ws_size;
}
